// Attention_53747220742202
// MI455X (gfx1250) — matrix operand order repaired and checked by reading, not compiled, not run on hardware
//
#include <hip/hip_runtime.h>
#include <hip/hip_bf16.h>

// ---------------------------------------------------------------------------
// Attention block for MI455X (gfx1250), bf16 WMMA everywhere.
// b=16, c=512, hw=1024, heads=4, dh=128, groups=32.
// GEMMs: 4x2 register-blocked WMMA tiles per wave (64x32 output / wave).
// ---------------------------------------------------------------------------

typedef __bf16 bf16;
typedef __attribute__((ext_vector_type(16))) __bf16 v16bf;
typedef __attribute__((ext_vector_type(8)))  __bf16 v8bf;
typedef __attribute__((ext_vector_type(8)))  float  v8f;

#define B_      16
#define C_      512
#define HW_     1024
#define HEADS_  4
#define DH_     128
#define GROUPS_ 32
#define EPS_    1e-5f
// scale = c^-0.5 folded with log2(e) so softmax uses exp2 (v_exp_f32)
#define QSCALE_ 0.06375871f

static __device__ __forceinline__ v8f wmma_bf16(v16bf a, v16bf b, v8f c) {
    return __builtin_amdgcn_wmma_f32_16x16x32_bf16(false, a, false, b,
                                                   (short)0, c, false, false);
}

// Load a 16x32 bf16 A-operand fragment (contiguous k = 16*lhalf + s) from a
// row-major matrix with row stride `ldr`, row base `rowptr` = row(lane%16).
static __device__ __forceinline__ v16bf load_afrag(const bf16* rowptr, int k0,
                                                   int lhalf) {
    v16bf af;
    *(v8bf*)&af       = *(const v8bf*)(rowptr + k0 + (lhalf ? 16 : 0));
    *((v8bf*)&af + 1) = *(const v8bf*)(rowptr + k0 + (lhalf ? 24 : 8));
    return af;
}

// ---------------------------------------------------------------------------
// Kernel 1: GroupNorm. One workgroup per (batch, group).
// Writes xn fp32 (b,c,hw) for the residual, and xn_t bf16 (b,hw,c) which is
// directly the WMMA B-operand layout for the QKV GEMM (K=channel contiguous).
// ---------------------------------------------------------------------------
__global__ __launch_bounds__(256) void gn_kernel(
    const float* __restrict__ x, const float* __restrict__ w,
    const float* __restrict__ bi, float* __restrict__ xn,
    bf16* __restrict__ xn_t) {
    __shared__ float r1[256];
    __shared__ float r2[256];
    const int bg = blockIdx.x;
    const int b = bg >> 5, g = bg & 31;
    const float* base = x + ((size_t)b * C_ + g * 16) * HW_;
    const int t = threadIdx.x;

    float s1 = 0.f, s2 = 0.f;
    for (int e = t; e < 16 * HW_; e += 256) {
        float v = base[e];
        s1 += v;
        s2 += v * v;
    }
    r1[t] = s1; r2[t] = s2;
    __syncthreads();
    for (int off = 128; off > 0; off >>= 1) {
        if (t < off) { r1[t] += r1[t + off]; r2[t] += r2[t + off]; }
        __syncthreads();
    }
    const float mean = r1[0] * (1.0f / (16.0f * HW_));
    const float var  = r2[0] * (1.0f / (16.0f * HW_)) - mean * mean;
    const float rstd = rsqrtf(var + EPS_);

    for (int e = t; e < 16 * HW_; e += 256) {
        const int cl = e >> 10;          // 0..15
        const int n  = e & (HW_ - 1);
        const int c  = g * 16 + cl;
        const float y = (base[e] - mean) * rstd * w[c] + bi[c];
        xn[((size_t)b * C_ + c) * HW_ + n] = y;
        xn_t[((size_t)b * HW_ + n) * C_ + c] = (bf16)y;
    }
}

// ---------------------------------------------------------------------------
// Kernel 2: fp32 -> bf16 weight conversion (qkv_w 1536x512, proj_w 512x512).
// ---------------------------------------------------------------------------
__global__ __launch_bounds__(256) void cvt_kernel(
    const float* __restrict__ qw, const float* __restrict__ pw,
    bf16* __restrict__ qwb, bf16* __restrict__ pwb) {
    const int i = blockIdx.x * 256 + threadIdx.x;   // 0 .. 1048575
    if (i < 3 * C_ * C_) {
        qwb[i] = (bf16)qw[i];
    } else {
        const int j = i - 3 * C_ * C_;
        pwb[j] = (bf16)pw[j];
    }
}

// ---------------------------------------------------------------------------
// Kernel 3: QKV GEMM  qkv[o,n] = sum_c W[o,c]*xn_t[n,c] + bias, per batch.
// 8 waves/WG; each wave: 4 M-tiles x 2 N-tiles (64x32 output), K=512 in 16
// WMMA(k=32) steps -> 128 WMMAs/wave, A frags reused 2x, B frags 4x.
// Epilogue scatters into attention-ready layouts:
//   q_t (b,h,hw,dh) bf16 * QSCALE_   (A operand of S = Q K^T)
//   k_t (b,h,hw,dh) bf16             (B operand, K-run contiguous)
//   v_b (b,h,dh,hw) bf16             (B operand of O = P V^T)
// ---------------------------------------------------------------------------
__global__ __launch_bounds__(256) void qkv_gemm_kernel(
    const bf16* __restrict__ wq, const bf16* __restrict__ xn_t,
    const float* __restrict__ bias, bf16* __restrict__ q_t,
    bf16* __restrict__ k_t, bf16* __restrict__ v_b) {
    const int lane = threadIdx.x & 31, wave = threadIdx.x >> 5;
    const int lrow = lane & 15, lhalf = lane >> 4;
    const int n0 = blockIdx.x * 32;                 // 2 N-tiles
    const int m0 = (blockIdx.y * 8 + wave) * 64;    // 4 M-tiles, 0..1472
    const int b  = blockIdx.z;

    const bf16* arow = wq + (size_t)(m0 + lrow) * C_;
    const bf16* brow = xn_t + ((size_t)b * HW_ + n0 + lrow) * C_;

    v8f acc[4][2];
    #pragma unroll
    for (int mi = 0; mi < 4; ++mi)
        #pragma unroll
        for (int ni = 0; ni < 2; ++ni) acc[mi][ni] = (v8f){};

    for (int k0 = 0; k0 < C_; k0 += 32) {
        v16bf af[4], bfm[2];
        #pragma unroll
        for (int mi = 0; mi < 4; ++mi)
            af[mi] = load_afrag(arow + (size_t)mi * 16 * C_, k0, lhalf);
        #pragma unroll
        for (int ni = 0; ni < 2; ++ni)
            bfm[ni] = *(const v16bf*)(brow + (size_t)ni * 16 * C_ + k0 + lhalf * 16);
        #pragma unroll
        for (int mi = 0; mi < 4; ++mi)
            #pragma unroll
            for (int ni = 0; ni < 2; ++ni)
                acc[mi][ni] = wmma_bf16(af[mi], bfm[ni], acc[mi][ni]);
    }

    const int which = m0 / C_;                 // 0=q 1=k 2=v (wave-uniform)
    #pragma unroll
    for (int mi = 0; mi < 4; ++mi) {
        const int obase  = m0 + mi * 16 + 8 * lhalf;  // 8 consecutive rows
        const int olocal = obase - which * C_;        // 0..511
        const int h  = olocal >> 7;
        const int c0 = olocal & (DH_ - 1);
        const size_t bh = (size_t)(b * HEADS_ + h);
        #pragma unroll
        for (int ni = 0; ni < 2; ++ni) {
            const int n = n0 + ni * 16 + lrow;
            float vals[8];
            #pragma unroll
            for (int r = 0; r < 8; ++r) vals[r] = acc[mi][ni][r] + bias[obase + r];

            if (which == 0) {
                v8bf pk;
                #pragma unroll
                for (int r = 0; r < 8; ++r) pk[r] = (bf16)(vals[r] * QSCALE_);
                *(v8bf*)(q_t + (bh * HW_ + n) * DH_ + c0) = pk;
            } else if (which == 1) {
                v8bf pk;
                #pragma unroll
                for (int r = 0; r < 8; ++r) pk[r] = (bf16)vals[r];
                *(v8bf*)(k_t + (bh * HW_ + n) * DH_ + c0) = pk;
            } else {
                #pragma unroll
                for (int r = 0; r < 8; ++r)
                    v_b[(bh * DH_ + c0 + r) * HW_ + n] = (bf16)vals[r];
            }
        }
    }
}

// ---------------------------------------------------------------------------
// Kernel 4: flash attention. One wave per 16-query block of one (b,head).
// Online softmax (exp2 domain, scale folded into q). P transposed C->A layout
// through per-wave LDS (s_wait_dscnt). 16 WMMAs per 32-key step, 32 steps.
// Next step's K/V tiles pulled in with global_prefetch_b8.
// Output: attn-out transposed (b,hw,c) bf16 = B operand for proj GEMM.
// ---------------------------------------------------------------------------
__global__ __launch_bounds__(256) void attn_kernel(
    const bf16* __restrict__ q_t, const bf16* __restrict__ k_t,
    const bf16* __restrict__ v_b, bf16* __restrict__ ao_t) {
    __shared__ __align__(128) bf16 pbuf[8][16 * 32];

    const int lane = threadIdx.x & 31, wave = threadIdx.x >> 5;
    const int lrow = lane & 15, lhalf = lane >> 4;
    const int gw = blockIdx.x * 8 + wave;       // 0..4095
    const int qblk = gw & 63;
    const int bh = gw >> 6;                     // b*4 + h
    const int i0 = qblk * 16;

    const bf16* qbase = q_t + (size_t)bh * HW_ * DH_;
    const bf16* kbase = k_t + (size_t)bh * HW_ * DH_;
    const bf16* vbase = v_b + (size_t)bh * DH_ * HW_;

    // Q fragments: 4 chunks over dh=128 (A-operand layout)
    v16bf qf[4];
    const bf16* qrow = qbase + (size_t)(i0 + lrow) * DH_;
    #pragma unroll
    for (int kk = 0; kk < 4; ++kk)
        qf[kk] = load_afrag(qrow, kk * 32, lhalf);

    v8f o[8];
    #pragma unroll
    for (int t = 0; t < 8; ++t) o[t] = (v8f){};
    float m[8], lsum[8];
    #pragma unroll
    for (int r = 0; r < 8; ++r) { m[r] = -1.0e30f; lsum[r] = 0.f; }

    bf16* pb = pbuf[wave];

    for (int j0 = 0; j0 < HW_; j0 += 32) {
        // ---- prefetch next 32-key step (K rows + V row segments) ----
        if (j0 + 32 < HW_) {
            __builtin_prefetch(kbase + (size_t)(j0 + 32 + lane) * DH_, 0, 0);
            #pragma unroll
            for (int pv = 0; pv < 4; ++pv)
                __builtin_prefetch(
                    vbase + (size_t)(lane * 4 + pv) * HW_ + j0 + 32, 0, 0);
        }

        // ---- S tiles: 16 queries x 32 keys, K-dim = 128 ----
        v8f s0 = {}, s1 = {};
        #pragma unroll
        for (int kk = 0; kk < 4; ++kk) {
            const int kb = kk * 32 + lhalf * 16;
            const v16bf kf0 = *(const v16bf*)(kbase + (size_t)(j0 + lrow) * DH_ + kb);
            const v16bf kf1 = *(const v16bf*)(kbase + (size_t)(j0 + 16 + lrow) * DH_ + kb);
            s0 = wmma_bf16(qf[kk], kf0, s0);
            s1 = wmma_bf16(qf[kk], kf1, s1);
        }

        // ---- online softmax (base-2 domain) ----
        float rm[8];
        #pragma unroll
        for (int r = 0; r < 8; ++r) rm[r] = fmaxf(s0[r], s1[r]);
        #pragma unroll
        for (int xm = 1; xm <= 8; xm <<= 1)
            #pragma unroll
            for (int r = 0; r < 8; ++r)
                rm[r] = fmaxf(rm[r], __shfl_xor(rm[r], xm, 32));

        float alpha[8];
        #pragma unroll
        for (int r = 0; r < 8; ++r) {
            const float mn = fmaxf(m[r], rm[r]);
            alpha[r] = __builtin_exp2f(m[r] - mn);
            m[r] = mn;
        }
        #pragma unroll
        for (int r = 0; r < 8; ++r) {
            s0[r] = __builtin_exp2f(s0[r] - m[r]);
            s1[r] = __builtin_exp2f(s1[r] - m[r]);
        }
        float rs[8];
        #pragma unroll
        for (int r = 0; r < 8; ++r) rs[r] = s0[r] + s1[r];
        #pragma unroll
        for (int xm = 1; xm <= 8; xm <<= 1)
            #pragma unroll
            for (int r = 0; r < 8; ++r)
                rs[r] += __shfl_xor(rs[r], xm, 32);
        #pragma unroll
        for (int r = 0; r < 8; ++r) lsum[r] = lsum[r] * alpha[r] + rs[r];
        #pragma unroll
        for (int t = 0; t < 8; ++t)
            #pragma unroll
            for (int r = 0; r < 8; ++r) o[t][r] *= alpha[r];

        // ---- transpose P (C layout -> A layout) through LDS ----
        #pragma unroll
        for (int r = 0; r < 8; ++r) {
            const int row = r + 8 * lhalf;
            pb[row * 32 + lrow]      = (bf16)s0[r];
            pb[row * 32 + 16 + lrow] = (bf16)s1[r];
        }
        asm volatile("s_wait_dscnt 0x0" ::: "memory");
        v16bf pf = load_afrag(pb + lrow * 32, 0, lhalf);

        // ---- O += P * V^T (dh=128 => 8 column tiles) ----
        #pragma unroll
        for (int ct = 0; ct < 8; ++ct) {
            const v16bf vf = *(const v16bf*)(
                vbase + (size_t)(ct * 16 + lrow) * HW_ + j0 + lhalf * 16);
            o[ct] = wmma_bf16(pf, vf, o[ct]);
        }
    }

    // ---- finalize: O /= l, write transposed attn-out (b,hw,c512) ----
    float inv[8];
    #pragma unroll
    for (int r = 0; r < 8; ++r) inv[r] = 1.0f / lsum[r];

    const int b = bh >> 2, h = bh & 3;
    #pragma unroll
    for (int ct = 0; ct < 8; ++ct) {
        const int c = h * DH_ + ct * 16 + lrow;
        #pragma unroll
        for (int r = 0; r < 8; ++r) {
            const int i = i0 + r + 8 * lhalf;
            ao_t[((size_t)b * HW_ + i) * C_ + c] = (bf16)(o[ct][r] * inv[r]);
        }
    }
}

// ---------------------------------------------------------------------------
// Kernel 5: proj GEMM + bias + residual(xn), fp32 output (b,c,hw).
// 4x2 blocked like the QKV GEMM; 8 waves cover the full M=512 strip.
// ---------------------------------------------------------------------------
__global__ __launch_bounds__(256) void proj_gemm_kernel(
    const bf16* __restrict__ wp, const bf16* __restrict__ ao_t,
    const float* __restrict__ bias, const float* __restrict__ xn,
    float* __restrict__ out) {
    const int lane = threadIdx.x & 31, wave = threadIdx.x >> 5;
    const int lrow = lane & 15, lhalf = lane >> 4;
    const int n0 = blockIdx.x * 32;
    const int m0 = wave * 64;                  // 8 waves x 64 = 512 = C_
    const int b  = blockIdx.z;

    const bf16* arow = wp + (size_t)(m0 + lrow) * C_;
    const bf16* brow = ao_t + ((size_t)b * HW_ + n0 + lrow) * C_;

    v8f acc[4][2];
    #pragma unroll
    for (int mi = 0; mi < 4; ++mi)
        #pragma unroll
        for (int ni = 0; ni < 2; ++ni) acc[mi][ni] = (v8f){};

    for (int k0 = 0; k0 < C_; k0 += 32) {
        v16bf af[4], bfm[2];
        #pragma unroll
        for (int mi = 0; mi < 4; ++mi)
            af[mi] = load_afrag(arow + (size_t)mi * 16 * C_, k0, lhalf);
        #pragma unroll
        for (int ni = 0; ni < 2; ++ni)
            bfm[ni] = *(const v16bf*)(brow + (size_t)ni * 16 * C_ + k0 + lhalf * 16);
        #pragma unroll
        for (int mi = 0; mi < 4; ++mi)
            #pragma unroll
            for (int ni = 0; ni < 2; ++ni)
                acc[mi][ni] = wmma_bf16(af[mi], bfm[ni], acc[mi][ni]);
    }

    #pragma unroll
    for (int mi = 0; mi < 4; ++mi) {
        const int obase = m0 + mi * 16 + 8 * lhalf;
        #pragma unroll
        for (int ni = 0; ni < 2; ++ni) {
            const int n = n0 + ni * 16 + lrow;
            #pragma unroll
            for (int r = 0; r < 8; ++r) {
                const int oc = obase + r;
                const size_t idx = ((size_t)b * C_ + oc) * HW_ + n;
                out[idx] = acc[mi][ni][r] + bias[oc] + xn[idx];
            }
        }
    }
}

// ---------------------------------------------------------------------------
// Host launcher
// ---------------------------------------------------------------------------
extern "C" void kernel_launch(void* const* d_in, const int* in_sizes, int n_in,
                              void* d_out, int out_size, void* d_ws, size_t ws_size,
                              hipStream_t stream) {
    const float* x      = (const float*)d_in[0];
    const float* norm_w = (const float*)d_in[1];
    const float* norm_b = (const float*)d_in[2];
    const float* qkv_w  = (const float*)d_in[3];
    const float* qkv_b  = (const float*)d_in[4];
    const float* proj_w = (const float*)d_in[5];
    const float* proj_b = (const float*)d_in[6];
    float* out = (float*)d_out;

    char* ws = (char*)d_ws;
    size_t off = 0;
    float* xn_f32 = (float*)(ws + off); off += (size_t)B_ * C_ * HW_ * 4;       // 32 MB
    bf16*  xn_t   = (bf16*)(ws + off);  off += (size_t)B_ * HW_ * C_ * 2;       // 16 MB
    bf16*  qwb    = (bf16*)(ws + off);  off += (size_t)3 * C_ * C_ * 2;         // 1.5 MB
    bf16*  pwb    = (bf16*)(ws + off);  off += (size_t)C_ * C_ * 2;             // 0.5 MB
    bf16*  q_t    = (bf16*)(ws + off);  off += (size_t)B_ * HEADS_ * HW_ * DH_ * 2;
    bf16*  k_t    = (bf16*)(ws + off);  off += (size_t)B_ * HEADS_ * HW_ * DH_ * 2;
    bf16*  v_b    = (bf16*)(ws + off);  off += (size_t)B_ * HEADS_ * DH_ * HW_ * 2;
    bf16*  ao_t   = (bf16*)(ws + off);  off += (size_t)B_ * HW_ * C_ * 2;

    gn_kernel<<<B_ * GROUPS_, 256, 0, stream>>>(x, norm_w, norm_b, xn_f32, xn_t);
    cvt_kernel<<<(3 * C_ * C_ + C_ * C_) / 256, 256, 0, stream>>>(qkv_w, proj_w, qwb, pwb);
    // M=1536 -> 96 tiles -> 24 wave-slots of 4 tiles -> gridDim.y = 3
    qkv_gemm_kernel<<<dim3(HW_ / 32, 3, B_), 256, 0, stream>>>(
        qwb, xn_t, qkv_b, q_t, k_t, v_b);
    attn_kernel<<<(B_ * HEADS_ * (HW_ / 16)) / 8, 256, 0, stream>>>(q_t, k_t, v_b, ao_t);
    proj_gemm_kernel<<<dim3(HW_ / 32, 1, B_), 256, 0, stream>>>(
        pwb, ao_t, proj_b, xn_f32, out);
}
